// SRPLayer_4466765988635
// MI455X (gfx1250) — matrix operand order repaired and checked by reading, not compiled, not run on hardware
//
#include <hip/hip_runtime.h>

// SRP layer for MI455X (gfx1250), wave32 + WMMA bf16 (f32 accumulate, hi/lo split).
//
//   mean[o] = (1/n) sum_n Psi[o,n];   sx[d] = sum_n x[d,n]
//   pxt[o,d] = sum_n Psi[o,n] x[d,n]  - mean[o]*sx[d]      (128 x 512)
//   out[o,n] = sum_d pxt[o,d] x[d,n]                        (128 x 32768)

typedef __attribute__((ext_vector_type(16))) __bf16          v16bf;
typedef __attribute__((ext_vector_type(16))) unsigned short  v16u;
typedef __attribute__((ext_vector_type(8)))  float           v8f;

#define DFEAT 512
#define NCOL  32768
#define ODIM  128

// ---------------- bf16 split helpers (RNE via bit ops) ----------------
__device__ __forceinline__ unsigned short f2bf(float f) {
  unsigned u = __float_as_uint(f);
  return (unsigned short)((u + 0x7FFFu + ((u >> 16) & 1u)) >> 16);
}
__device__ __forceinline__ float bf2f(unsigned short h) {
  return __uint_as_float(((unsigned)h) << 16);
}
__device__ __forceinline__ void split2(float f, unsigned short& h, unsigned short& l) {
  h = f2bf(f);
  l = f2bf(f - bf2f(h));
}

// Load a 16-element bf16 A-operand slice for this lane from a row-major
// ushort matrix: elems[0..15] = K{kb+2*koff .. +15} (contiguous).
struct U32x8 { uint4 a, b; };
__device__ __forceinline__ v16bf load_a_bf16(const unsigned short* __restrict__ base,
                                             size_t rowoff, int kb, int koff) {
  const unsigned short* p = base + rowoff + kb + 2 * koff;
  U32x8 t;
  t.a = *(const uint4*)(p);        // 8 shorts: K = kb+2*koff+0..7
  t.b = *(const uint4*)(p + 8);    // 8 shorts: K = kb+2*koff+8..15
  return __builtin_bit_cast(v16bf, t);
}

// ---------------- kernel 1: zero pxt scratch (split-K atomics target) ----
__global__ void zero_pxt_kernel(float* __restrict__ pxt) {
  pxt[blockIdx.x * 256 + threadIdx.x] = 0.0f;
}

// ---------------- kernel 2: pre-split Psi into bf16 hi/lo ----------------
__global__ void convert_psi_kernel(const float* __restrict__ src,
                                   unsigned short* __restrict__ hi,
                                   unsigned short* __restrict__ lo) {
  int i = blockIdx.x * 256 + threadIdx.x;   // grid covers ODIM*NCOL exactly
  unsigned short h, l;
  split2(src[i], h, l);
  hi[i] = h; lo[i] = l;
}

// ---------------- kernel 3: row sums (mean of Psi rows, colsum of x rows) --
__global__ void rowsum_kernel(const float* __restrict__ x,
                              const float* __restrict__ Psi,
                              float* __restrict__ mean, float* __restrict__ sx) {
  __shared__ float red[256];
  const int b = blockIdx.x;                       // 0..639
  const float* row = (b < ODIM) ? (Psi + (size_t)b * NCOL)
                                : (x + (size_t)(b - ODIM) * NCOL);
  float s = 0.0f;
  for (int i = threadIdx.x; i < NCOL / 4; i += 256) {
    float4 v = ((const float4*)row)[i];
    s += v.x + v.y + v.z + v.w;
  }
  red[threadIdx.x] = s;
  __syncthreads();
  for (int o = 128; o > 0; o >>= 1) {
    if ((int)threadIdx.x < o) red[threadIdx.x] += red[threadIdx.x + o];
    __syncthreads();
  }
  if (threadIdx.x == 0) {
    if (b < ODIM) mean[b] = red[0] * (1.0f / NCOL);
    else          sx[b - ODIM] = red[0];
  }
}

// ---------------- kernel 4: GEMM1  pxt += Psi * x^T  (split-K, atomics) ----
// Grid: 32 d-tiles x 64 k-chunks = 2048 one-wave blocks. Each block covers
// all 8 o-tiles so each x chunk is converted to bf16 hi/lo exactly once.
__global__ void __launch_bounds__(32)
gemm1_kernel(const unsigned short* __restrict__ psi_hi,
             const unsigned short* __restrict__ psi_lo,
             const float* __restrict__ x, float* __restrict__ pxt) {
  const int lane = threadIdx.x;
  const int half = lane >> 4;         // 0: lanes 0-15, 1: lanes 16-31
  const int r16  = lane & 15;
  const int koff = half * 8;          // A-operand K sub-chunk select
  const int dt = blockIdx.x & 31;     // d-tile 0..31   (N dimension)
  const int kc = blockIdx.x >> 5;     // k-chunk 0..63  (split-K)
  const int d0 = dt * 16;
  const int kbase0 = kc * (NCOL / 64);        // 512 elements per chunk

  v8f acc[8];
#pragma unroll
  for (int ot = 0; ot < 8; ++ot) acc[ot] = 0.0f;

  const float* xrow = x + (size_t)(d0 + r16) * NCOL;   // B: N(col)=d via lane

#pragma unroll 1
  for (int ks = 0; ks < 16; ++ks) {
    const int kb = kbase0 + ks * 32;
    // ---- B operand from x: lanes 0-15 hold K 0-15, lanes 16-31 hold K 16-31
    float bv[16];
    {
      const float* p = xrow + kb + half * 16;
      const float4 q0 = *(const float4*)(p);
      const float4 q1 = *(const float4*)(p + 4);
      const float4 q2 = *(const float4*)(p + 8);
      const float4 q3 = *(const float4*)(p + 12);
      bv[0]=q0.x; bv[1]=q0.y; bv[2]=q0.z; bv[3]=q0.w;
      bv[4]=q1.x; bv[5]=q1.y; bv[6]=q1.z; bv[7]=q1.w;
      bv[8]=q2.x; bv[9]=q2.y; bv[10]=q2.z; bv[11]=q2.w;
      bv[12]=q3.x; bv[13]=q3.y; bv[14]=q3.z; bv[15]=q3.w;
    }
    v16u bhU, blU;
#pragma unroll
    for (int i = 0; i < 16; ++i) {
      unsigned short h, l; split2(bv[i], h, l); bhU[i] = h; blU[i] = l;
    }
    const v16bf bh = __builtin_bit_cast(v16bf, bhU);
    const v16bf bl = __builtin_bit_cast(v16bf, blU);

#pragma unroll
    for (int ot = 0; ot < 8; ++ot) {
      const size_t rowoff = (size_t)(ot * 16 + r16) * NCOL;
      const v16bf ah = load_a_bf16(psi_hi, rowoff, kb, koff);
      const v16bf al = load_a_bf16(psi_lo, rowoff, kb, koff);
      acc[ot] = __builtin_amdgcn_wmma_f32_16x16x32_bf16(false, ah, false, bh,
                                                        (short)0, acc[ot], false, false);
      acc[ot] = __builtin_amdgcn_wmma_f32_16x16x32_bf16(false, ah, false, bl,
                                                        (short)0, acc[ot], false, false);
      acc[ot] = __builtin_amdgcn_wmma_f32_16x16x32_bf16(false, al, false, bh,
                                                        (short)0, acc[ot], false, false);
    }
  }

  // C/D layout: VGPR r -> M = r (lanes 0-15) or 8+r (lanes 16-31), N = r16
#pragma unroll
  for (int ot = 0; ot < 8; ++ot) {
#pragma unroll
    for (int r = 0; r < 8; ++r) {
      const int orow = ot * 16 + half * 8 + r;
      atomicAdd(&pxt[orow * DFEAT + d0 + r16], acc[ot][r]);
    }
  }
}

// ---------------- kernel 5: centering fixup + pre-split pxt to bf16 -------
__global__ void fixup_kernel(const float* __restrict__ pxt,
                             const float* __restrict__ mean,
                             const float* __restrict__ sx,
                             unsigned short* __restrict__ pxt_hi,
                             unsigned short* __restrict__ pxt_lo) {
  const int i = blockIdx.x * 256 + threadIdx.x;   // 65536 elements
  const int o = i >> 9, d = i & 511;
  const float v = pxt[i] - mean[o] * sx[d];
  unsigned short h, l;
  split2(v, h, l);
  pxt_hi[i] = h; pxt_lo[i] = l;
}

// ---------------- kernel 6: GEMM2  out = pxt_c * x  ------------------------
// Grid: 2048 n-tiles, one wave each; each wave produces a full 128x16 stripe
// (8 o-tile accumulators) so x columns are split to bf16 once per stripe.
__global__ void __launch_bounds__(32)
gemm2_kernel(const unsigned short* __restrict__ pxt_hi,
             const unsigned short* __restrict__ pxt_lo,
             const float* __restrict__ x, float* __restrict__ out) {
  const int lane = threadIdx.x;
  const int half = lane >> 4;
  const int r16  = lane & 15;
  const int koff = half * 8;
  const int n0   = blockIdx.x * 16;
  const int col  = n0 + r16;

  v8f acc[8];
#pragma unroll
  for (int ot = 0; ot < 8; ++ot) acc[ot] = 0.0f;

#pragma unroll 1
  for (int ks = 0; ks < 16; ++ks) {          // K = DFEAT = 512, steps of 32
    const int kb = ks * 32;
    // ---- B operand from x columns: elem i -> K = kb + half*16 + i
    float bv[16];
#pragma unroll
    for (int i = 0; i < 16; ++i)
      bv[i] = x[(size_t)(kb + half * 16 + i) * NCOL + col];
    v16u bhU, blU;
#pragma unroll
    for (int i = 0; i < 16; ++i) {
      unsigned short h, l; split2(bv[i], h, l); bhU[i] = h; blU[i] = l;
    }
    const v16bf bh = __builtin_bit_cast(v16bf, bhU);
    const v16bf bl = __builtin_bit_cast(v16bf, blU);

#pragma unroll
    for (int ot = 0; ot < 8; ++ot) {
      const size_t rowoff = (size_t)(ot * 16 + r16) * DFEAT;
      const v16bf ah = load_a_bf16(pxt_hi, rowoff, kb, koff);
      const v16bf al = load_a_bf16(pxt_lo, rowoff, kb, koff);
      acc[ot] = __builtin_amdgcn_wmma_f32_16x16x32_bf16(false, ah, false, bh,
                                                        (short)0, acc[ot], false, false);
      acc[ot] = __builtin_amdgcn_wmma_f32_16x16x32_bf16(false, ah, false, bl,
                                                        (short)0, acc[ot], false, false);
      acc[ot] = __builtin_amdgcn_wmma_f32_16x16x32_bf16(false, al, false, bh,
                                                        (short)0, acc[ot], false, false);
    }
  }

#pragma unroll
  for (int ot = 0; ot < 8; ++ot) {
#pragma unroll
    for (int r = 0; r < 8; ++r) {
      const int orow = ot * 16 + half * 8 + r;
      out[(size_t)orow * NCOL + col] = acc[ot][r];
    }
  }
}

// ---------------- host-side launch -----------------------------------------
extern "C" void kernel_launch(void* const* d_in, const int* in_sizes, int n_in,
                              void* d_out, int out_size, void* d_ws, size_t ws_size,
                              hipStream_t stream) {
  const float* x   = (const float*)d_in[0];   // (512, 32768)
  const float* Psi = (const float*)d_in[1];   // (128, 32768)
  float* out = (float*)d_out;                 // (128, 32768)

  // Workspace layout (f32 region then ushort region; all 16B aligned):
  float* ws    = (float*)d_ws;
  float* pxt   = ws;                          // 65536 f32
  float* mean  = ws + 65536;                  // 128 f32
  float* sx    = mean + 128;                  // 512 f32
  unsigned short* psi_hi = (unsigned short*)(ws + 66176);
  unsigned short* psi_lo = psi_hi + (size_t)ODIM * NCOL;
  unsigned short* pxt_hi = psi_lo + (size_t)ODIM * NCOL;
  unsigned short* pxt_lo = pxt_hi + (size_t)ODIM * DFEAT;

  zero_pxt_kernel<<<256, 256, 0, stream>>>(pxt);
  convert_psi_kernel<<<(ODIM * NCOL) / 256, 256, 0, stream>>>(Psi, psi_hi, psi_lo);
  rowsum_kernel<<<ODIM + DFEAT, 256, 0, stream>>>(x, Psi, mean, sx);
  gemm1_kernel<<<32 * 64, 32, 0, stream>>>(psi_hi, psi_lo, x, pxt);
  fixup_kernel<<<(ODIM * DFEAT) / 256, 256, 0, stream>>>(pxt, mean, sx, pxt_hi, pxt_lo);
  gemm2_kernel<<<NCOL / 16, 32, 0, stream>>>(pxt_hi, pxt_lo, x, out);
}
